// KANSplineLayer_88991722373383
// MI455X (gfx1250) — matrix operand order repaired and checked by reading, not compiled, not run on hardware
//
#include <hip/hip_runtime.h>
#include <hip/hip_bf16.h>
#include <stdint.h>

typedef __attribute__((ext_vector_type(16))) _Float16 v16h;
typedef __attribute__((ext_vector_type(8)))  _Float16 v8h;
typedef __attribute__((ext_vector_type(4)))  _Float16 v4h;
typedef __attribute__((ext_vector_type(8)))  float    v8f;
typedef __attribute__((ext_vector_type(4)))  float    v4f;
typedef __attribute__((ext_vector_type(4)))  unsigned v4u;
typedef __attribute__((ext_vector_type(8)))  int      v8i;
typedef __attribute__((ext_vector_type(4)))  int      v4i;

#if __has_builtin(__builtin_amdgcn_tensor_load_to_lds) && __has_builtin(__builtin_amdgcn_s_wait_tensorcnt)
#define USE_TDM 1
#else
#define USE_TDM 0
#endif

#define IN_F   256
#define OUT_F  256
#define NROWS  32768
#define KKNOT  9
#define SPK    (IN_F * KKNOT)   // 2304

// workspace layout (bytes)
#define WS_OFF   0                      // 256 f32 (min)
#define WS_SCL   1024                   // 256 f32 (1/range)
#define WS_RNG   2048                   // 256 f32 (range)
#define WS_BCOL  3072                   // 256 f32 (min @ Wb)
#define WS_MINU  4096                   // 256 u32
#define WS_MAXU  5120                   // 256 u32
#define WS_BNP   6144                   // 256 float4 (s1, o1', o2, 1/s2)
#define WS_S2    10240                  // 256 f32 (s2)
#define WS_WB    11264                  // 256*256 f16, [o][k]  (range-scaled)
#define WS_WSP   (11264 + 131072)       // 256*2304 f16, [o][kidx]

union V16 { v16h v; v8h h[2]; };
union V8F { v8f v; v4f q[2]; };

// ---- order-preserving float<->uint encoding for atomic min/max ----
__device__ inline unsigned fenc(float f) {
    unsigned b = __float_as_uint(f);
    return (b & 0x80000000u) ? ~b : (b | 0x80000000u);
}
__device__ inline float fdec(unsigned u) {
    unsigned b = (u & 0x80000000u) ? (u & 0x7FFFFFFFu) : ~u;
    return __uint_as_float(b);
}

__global__ __launch_bounds__(256) void k_init(unsigned* minU, unsigned* maxU) {
    int t = threadIdx.x;
    minU[t] = 0xFFFFFFFFu;
    maxU[t] = 0u;
}

__global__ __launch_bounds__(256) void k_minmax(const float* __restrict__ xg,
                                                unsigned* minU, unsigned* maxU) {
    int t = threadIdx.x;
    size_t r0 = (size_t)blockIdx.x * 128;
    float mn = 3.0e38f, mx = -3.0e38f;
    for (int r = 0; r < 128; ++r) {
        float v = xg[(r0 + r) * IN_F + t];
        mn = fminf(mn, v);
        mx = fmaxf(mx, v);
    }
    atomicMin(&minU[t], fenc(mn));
    atomicMax(&maxU[t], fenc(mx));
}

__global__ __launch_bounds__(256) void k_fin(const unsigned* minU, const unsigned* maxU,
                                             float* offp, float* sclp, float* rngp) {
    int t = threadIdx.x;
    float mn = fdec(minU[t]);
    float mx = fdec(maxU[t]);
    float r = mx - mn + 1e-7f;
    offp[t] = mn;
    rngp[t] = r;
    sclp[t] = 1.0f / r;
}

// bcol[o] = sum_k min[k] * Wb[k][o]
__global__ __launch_bounds__(256) void k_basecol(const float* __restrict__ bw,
                                                 const float* __restrict__ offp,
                                                 float* __restrict__ bcol) {
    int o = threadIdx.x;
    float s = 0.0f;
    for (int k = 0; k < IN_F; ++k) s += offp[k] * bw[k * OUT_F + o];
    bcol[o] = s;
}

// pack per-column BN constants: (s1, o1+s1*bcol, o2, 1/s2) + s2
__global__ __launch_bounds__(256) void k_bnprep(
    const float* __restrict__ bg, const float* __restrict__ bb,
    const float* __restrict__ bm, const float* __restrict__ bv,
    const float* __restrict__ sg, const float* __restrict__ sb,
    const float* __restrict__ sm, const float* __restrict__ sv,
    const float* __restrict__ bcol,
    float* __restrict__ bnp, float* __restrict__ s2p) {
    int c = threadIdx.x;
    float s1 = bg[c] * rsqrtf(bv[c] + 1e-3f);
    float o1 = bb[c] - bm[c] * s1 + s1 * bcol[c];
    float s2 = sg[c] * rsqrtf(sv[c] + 1e-3f);
    float o2 = sb[c] - sm[c] * s2;
    bnp[c * 4 + 0] = s1;
    bnp[c * 4 + 1] = o1;
    bnp[c * 4 + 2] = o2;
    bnp[c * 4 + 3] = 1.0f / s2;
    s2p[c] = s2;
}

// wbT[o][k] = f16(range[k] * Wb[k][o])  (base GEMM runs on xn)
__global__ __launch_bounds__(256) void k_cvt_base(const float* __restrict__ bw,
                                                  const float* __restrict__ rngp,
                                                  _Float16* __restrict__ wbT) {
    int i = blockIdx.x * 256 + threadIdx.x;   // 65536 total
    int o = i >> 8, k = i & 255;
    wbT[i] = (_Float16)(rngp[k] * bw[k * OUT_F + o]);
}

// wspT[o][kidx] = f16(spline_weight[o][in][kk] * scaler[o][in]), kidx = in*9+kk
__global__ __launch_bounds__(256) void k_cvt_spline(const float* __restrict__ sw,
                                                    const float* __restrict__ sc,
                                                    _Float16* __restrict__ wspT) {
    int kidx = blockIdx.x * 256 + threadIdx.x;   // grid.x = 9 -> 2304
    int o = blockIdx.y;
    int in_ = kidx / 9;
    float v = sw[(size_t)o * SPK + kidx] * sc[o * IN_F + in_];
    wspT[(size_t)o * SPK + kidx] = (_Float16)v;
}

#define ROWS 32     // rows per block
#define XPAD 264    // padded row stride (halves) for xn
#define BST  40     // padded per-column / per-row stride (halves) for chunks
#define OST  36     // padded per-column stride (floats) for output tile

#if USE_TDM
// One TDM descriptor moves a 32(K) x 256(col) f16 tile, K-contiguous per column,
// into LDS with 16B padding after every 64B row -> matches BST=40 layout.
__device__ inline void tdm_stage(const _Float16* W, unsigned rstride, int kbase,
                                 _Float16* dst) {
    unsigned long long ga = (unsigned long long)(uintptr_t)(W + kbase);
    unsigned la = (unsigned)(uintptr_t)dst;   // low 32 bits of LDS flat addr
    v4u g0;
    g0[0] = 1u;                                     // count=1, user descriptor
    g0[1] = la;                                     // lds_addr
    g0[2] = (unsigned)ga;                           // global_addr[31:0]
    g0[3] = ((unsigned)(ga >> 32) & 0x01FFFFFFu) | 0x80000000u;  // addr[56:32] | type=2
    v8i g1;
    g1[0] = (int)((1u << 16)      // data_size = 2B
                | (1u << 20)      // pad_enable
                | (3u << 22)      // pad_interval: every 16 dwords (64B)
                | (3u << 25));    // pad_amount: 4 dwords (16B)
    g1[1] = (int)((rstride & 0xFFFFu) << 16);             // tensor_dim0[15:0]
    g1[2] = (int)((rstride >> 16) | (256u << 16));        // dim0[31:16] | dim1[15:0]=256
    g1[3] = (int)(32u << 16);                             // dim1[31:16]=0 | tile_dim0=32
    g1[4] = 256;                                          // tile_dim1=256, tile_dim2=0
    g1[5] = (int)rstride;                                 // tensor_dim0_stride[31:0]
    g1[6] = 0;                                            // stride hi / dim1_stride
    g1[7] = 0;
    v4i z4 = {0, 0, 0, 0};
    v8i z8 = {0, 0, 0, 0, 0, 0, 0, 0};
    __builtin_amdgcn_tensor_load_to_lds(g0, g1, z4, z4, z8, 0);
}
#endif

__global__ __launch_bounds__(256) void k_kan_main(
    const float* __restrict__ xg,
    const _Float16* __restrict__ wbT,
    const _Float16* __restrict__ wspT,
    const float* __restrict__ offp, const float* __restrict__ sclp,
    const float* __restrict__ bnp, const float* __restrict__ s2p,
    float* __restrict__ outp)
{
    __shared__ __align__(16) _Float16 xnL[ROWS * XPAD];     // 16896 B
    __shared__ __align__(16) _Float16 Bs[2][256 * BST];     // 2 x 20480 B (reused as out tile)
    __shared__ __align__(16) _Float16 basT[2][ROWS * BST];  // 2 x 2560 B

    const int t = threadIdx.x;
    const size_t row0 = (size_t)blockIdx.x * ROWS;

    const int lane = t & 31, wv = t >> 5;
    const int mt = wv & 1;        // M tile (2 x 16 rows)
    const int nq = wv >> 1;       // N quarter (4 x 64 cols)
    const int mrow = lane & 15, khi = lane >> 4;
    const int arow = mt * 16 + mrow;
    const int col0 = nq * 64 + mrow;           // this lane's first column
    const int bofs = col0 * BST + khi * 16;    // LDS half-offset of first B frag

    // ---- stage normalized x into LDS (vectorized: 4 cols/thread, 8 iters) ----
    {
        const int colx = (t * 4) & 255;
        const int rsub = t >> 6;                 // 0..3
        v4f offv = *(const v4f*)(offp + colx);
        v4f sclv = *(const v4f*)(sclp + colx);
        #pragma unroll
        for (int i = 0; i < 8; ++i) {
            int row = i * 4 + rsub;
            v4f xv = *(const v4f*)(xg + (row0 + row) * IN_F + colx);
            v4h h;
            #pragma unroll
            for (int j = 0; j < 4; ++j) h[j] = (_Float16)((xv[j] - offv[j]) * sclv[j]);
            *(v4h*)(xnL + row * XPAD + colx) = h;
        }
    }

    // stage one 32-K weight chunk (all 256 cols) into LDS
    auto stage = [&](const _Float16* W, int rstride, int kbase, _Float16* dst) {
#if USE_TDM
        if (wv == 0) tdm_stage(W, (unsigned)rstride, kbase, dst);
#else
        const v8h* s = (const v8h*)(W + (size_t)t * rstride + kbase);
        v8h* d = (v8h*)(dst + t * BST);
        d[0] = s[0]; d[1] = s[1]; d[2] = s[2]; d[3] = s[3];
#endif
    };
    auto stageWait = [&]() {
#if USE_TDM
        if (wv == 0) __builtin_amdgcn_s_wait_tensorcnt((short)0);
#endif
    };

    // cooperative basis tile build: 32 rows x 32 kidx, 4 elements per thread
    auto buildBasis = [&](int kbase, _Float16* bt) {
        const int row = t >> 3;           // 0..31
        const int kws = (t & 7) * 4;      // 0,4,...,28
        const int kidx0 = kbase + kws;
        const int in0 = kidx0 / 9;
        const int kk0 = kidx0 - in0 * 9;
        float xnA = (float)xnL[row * XPAD + in0];
        int inB = in0 + 1; if (inB > IN_F - 1) inB = IN_F - 1;
        float xnB = (float)xnL[row * XPAD + inB];
        v4h r;
        #pragma unroll
        for (int j = 0; j < 4; ++j) {
            int kk = kk0 + j;
            int w = (kk >= KKNOT) ? 1 : 0;
            kk -= KKNOT * w;
            float xv = w ? xnB : xnA;
            float bas = fmaxf(0.0f, 1.0f - fabsf(xv - (-1.0f + 0.25f * (float)kk)));
            r[j] = (_Float16)bas;
        }
        *(v4h*)(bt + row * BST + kws) = r;
    };

    v8f acc[4];
    {
        v8f zero = {};
        #pragma unroll
        for (int tt = 0; tt < 4; ++tt) acc[tt] = zero;
    }
    // batch all fragment loads into distinct registers, then issue 4 WMMAs
    auto computeTiles = [&](const _Float16* ap, const _Float16* wbuf) {
        V16 a, b0, b1, b2, b3;
        const v8h* p = (const v8h*)ap;
        const v8h* q0 = (const v8h*)(wbuf + bofs);
        const v8h* q1 = (const v8h*)(wbuf + bofs + 16 * BST);
        const v8h* q2 = (const v8h*)(wbuf + bofs + 32 * BST);
        const v8h* q3 = (const v8h*)(wbuf + bofs + 48 * BST);
        a.h[0]  = p[0];  a.h[1]  = p[1];   // two contiguous 16B segs
        b0.h[0] = q0[0]; b0.h[1] = q0[1];
        b1.h[0] = q1[0]; b1.h[1] = q1[1];
        b2.h[0] = q2[0]; b2.h[1] = q2[1];
        b3.h[0] = q3[0]; b3.h[1] = q3[1];
        acc[0] = __builtin_amdgcn_wmma_f32_16x16x32_f16(
            false, a.v, false, b0.v, (short)0, acc[0], false, false);
        acc[1] = __builtin_amdgcn_wmma_f32_16x16x32_f16(
            false, a.v, false, b1.v, (short)0, acc[1], false, false);
        acc[2] = __builtin_amdgcn_wmma_f32_16x16x32_f16(
            false, a.v, false, b2.v, (short)0, acc[2], false, false);
        acc[3] = __builtin_amdgcn_wmma_f32_16x16x32_f16(
            false, a.v, false, b3.v, (short)0, acc[3], false, false);
    };

    // ================= base GEMM: xn @ (range*Wb), K = 256 =================
    stage(wbT, IN_F, 0, Bs[0]);
    for (int kc = 0; kc < 8; ++kc) {
        stageWait();                 // chunk kc's DMA complete (issued last iter)
        __syncthreads();
        if (kc < 7) stage(wbT, IN_F, (kc + 1) * 32, Bs[(kc + 1) & 1]);
        computeTiles(xnL + arow * XPAD + kc * 32 + khi * 16, Bs[kc & 1]);
    }

    // ---- mid transform: acc = (silu(bn_base(acc)) + o2) / s2 ----
    float s2v[4];
    #pragma unroll
    for (int tt = 0; tt < 4; ++tt) {
        const int col = col0 + tt * 16;
        v4f q = *(const v4f*)(bnp + col * 4);       // (s1, o1', o2, 1/s2)
        s2v[tt] = s2p[col];
        #pragma unroll
        for (int e = 0; e < 8; ++e) {
            float z = acc[tt][e] * q[0] + q[1];
            float r = z / (1.0f + __expf(-z));      // SiLU
            acc[tt][e] = (r + q[2]) * q[3];
        }
    }

    // ================= spline GEMM: K = 2304, cooperative basis tiles ========
    stage(wspT, SPK, 0, Bs[0]);
    buildBasis(0, basT[0]);
    for (int kc = 0; kc < 72; ++kc) {
        stageWait();
        __syncthreads();
        if (kc < 71) {
            stage(wspT, SPK, (kc + 1) * 32, Bs[(kc + 1) & 1]);
            buildBasis((kc + 1) * 32, basT[(kc + 1) & 1]);
        }
        computeTiles(basT[kc & 1] + arow * BST + khi * 16, Bs[kc & 1]);
    }

    // ---- epilogue: scale, transpose through LDS, coalesced b128 stores ----
    float* outT = (float*)(&Bs[0][0]);   // 256 cols x OST floats (36864 B <= 40960 B)
    __syncthreads();                     // last chunk's Bs/basT reads complete
    #pragma unroll
    for (int tt = 0; tt < 4; ++tt) {
        const int col = col0 + tt * 16;
        V8F u;
        #pragma unroll
        for (int e = 0; e < 8; ++e) u.v[e] = acc[tt][e] * s2v[tt];
        v4f* dst = (v4f*)(outT + col * OST + mt * 16 + khi * 8);
        dst[0] = u.q[0];
        dst[1] = u.q[1];
    }
    __syncthreads();
    {
        const int colx = (t * 4) & 255;
        const int rsub = t >> 6;                 // 0..3
        #pragma unroll
        for (int i = 0; i < 8; ++i) {
            int row = i * 4 + rsub;
            v4f o;
            #pragma unroll
            for (int j = 0; j < 4; ++j) o[j] = outT[(colx + j) * OST + row];
            *(v4f*)(outp + (row0 + row) * OUT_F + colx) = o;
        }
    }
}

extern "C" void kernel_launch(void* const* d_in, const int* in_sizes, int n_in,
                              void* d_out, int out_size, void* d_ws, size_t ws_size,
                              hipStream_t stream) {
    const float* xg = (const float*)d_in[0];
    const float* bw = (const float*)d_in[1];
    const float* sw = (const float*)d_in[2];
    const float* sc = (const float*)d_in[3];
    const float* bg = (const float*)d_in[4];
    const float* bb = (const float*)d_in[5];
    const float* bm = (const float*)d_in[6];
    const float* bv = (const float*)d_in[7];
    const float* sg = (const float*)d_in[8];
    const float* sb = (const float*)d_in[9];
    const float* sm = (const float*)d_in[10];
    const float* sv = (const float*)d_in[11];
    float* outp = (float*)d_out;

    char* ws = (char*)d_ws;
    float*    offp = (float*)(ws + WS_OFF);
    float*    sclp = (float*)(ws + WS_SCL);
    float*    rngp = (float*)(ws + WS_RNG);
    float*    bcol = (float*)(ws + WS_BCOL);
    unsigned* minU = (unsigned*)(ws + WS_MINU);
    unsigned* maxU = (unsigned*)(ws + WS_MAXU);
    float*    bnp  = (float*)(ws + WS_BNP);
    float*    s2p  = (float*)(ws + WS_S2);
    _Float16* wbT  = (_Float16*)(ws + WS_WB);
    _Float16* wspT = (_Float16*)(ws + WS_WSP);

    k_init<<<1, 256, 0, stream>>>(minU, maxU);
    k_minmax<<<256, 256, 0, stream>>>(xg, minU, maxU);
    k_fin<<<1, 256, 0, stream>>>(minU, maxU, offp, sclp, rngp);
    k_basecol<<<1, 256, 0, stream>>>(bw, offp, bcol);
    k_bnprep<<<1, 256, 0, stream>>>(bg, bb, bm, bv, sg, sb, sm, sv, bcol, bnp, s2p);
    k_cvt_base<<<256, 256, 0, stream>>>(bw, rngp, wbT);
    k_cvt_spline<<<dim3(9, 256), 256, 0, stream>>>(sw, sc, wspT);
    k_kan_main<<<NROWS / ROWS, 256, 0, stream>>>(
        xg, wbT, wspT, offp, sclp, bnp, s2p, outp);
}
